// ClinicalGatedGCN_70858370450162
// MI455X (gfx1250) — matrix operand order repaired and checked by reading, not compiled, not run on hardware
//
#include <hip/hip_runtime.h>
#include <hip/hip_bf16.h>

#define HDIM 128
#define NNODES 50000
#define NEDGES 800000
#define NGRAPH 64
#define NCLIN 16
#define NCLASS 2
#define MTILES (NNODES / 16)   // 3125, exact
#define HH (HDIM * HDIM)

#if defined(__has_builtin)
#if __has_builtin(__builtin_amdgcn_global_load_async_to_lds_b128)
#define HAVE_ASYNC_LDS 1
#endif
#endif

typedef __attribute__((ext_vector_type(16))) __bf16 v16bf;
typedef __attribute__((ext_vector_type(8)))  float  v8f;
typedef int v4i_vs __attribute__((vector_size(16)));   // matches builtin param type

union Frag16 {
    v16bf v;
    unsigned short u[16];
    uint4 q[2];
};

__device__ __forceinline__ unsigned short f2bf_rne(float f) {
    unsigned int u = __float_as_uint(f);
    unsigned int r = u + 0x7FFFu + ((u >> 16) & 1u);
    return (unsigned short)(r >> 16);
}

__device__ __forceinline__ float sigm(float x) {
    return 1.0f / (1.0f + __expf(-x));
}

// ---------------------------------------------------------------------------
// Convert fp32 activations -> bf16 (rounded) for WMMA input
// ---------------------------------------------------------------------------
__global__ void cvt_f32_bf16_kernel(const float* __restrict__ src,
                                    unsigned short* __restrict__ dst, int n) {
    int i = blockIdx.x * blockDim.x + threadIdx.x;
    if (i < n) dst[i] = f2bf_rne(src[i]);
}

// ---------------------------------------------------------------------------
// Convert all 3 layers x 4 weight matrices [H,H] fp32 -> bf16 AND pre-swizzle
// into per-lane WMMA B-fragment order:
//   Wb[l][w][frag=nt*4+kk][lane][i(0..15)]
// where for fragment (nt,kk), lane:
//   col  = nt*16 + (lane & 15)
//   krow = kk*32 + ((lane>>4)<<3) + (i&7) + ((i>>3)<<4)
// so each lane's 16 B-matrix halves are CONTIGUOUS (2 x b128 loads, zero
// packing movs in the GEMM hot loop).
// ---------------------------------------------------------------------------
__global__ void cvt_weights_kernel(const float* __restrict__ Wk,
                                   const float* __restrict__ Wq,
                                   const float* __restrict__ Wv,
                                   const float* __restrict__ Ws,
                                   unsigned short* __restrict__ Wb) {
    int o = blockIdx.x * blockDim.x + threadIdx.x;
    const int total = 3 * 4 * HH;
    if (o >= total) return;
    const int l = o / (4 * HH);
    const int w = (o / HH) & 3;
    const int r = o % HH;
    const int f    = r >> 9;          // fragment 0..31 (nt*4 + kk)
    const int nt   = f >> 2;
    const int kk   = f & 3;
    const int lane = (r >> 4) & 31;
    const int i    = r & 15;
    const int col  = nt * 16 + (lane & 15);
    const int krow = kk * 32 + ((lane >> 4) << 3) + (i & 7) + ((i >> 3) << 4);
    const float* srcs[4] = {Wk, Wq, Wv, Ws};
    Wb[o] = f2bf_rne(srcs[w][(size_t)l * HH + krow * HDIM + col]);
}

// ---------------------------------------------------------------------------
// Fused 4-way projection GEMM: out_y = hb @ W_y + b_y for y in {k,q,v,s}.
// blockIdx.y selects projection. Block stages the 32 KB pre-swizzled weight
// matrix in LDS (async global->LDS path when available); 8 waves/block, each
// wave computes a 16x128 output strip with K=128 = 4 x v_wmma bf16 per N-tile.
// B fragments: 2 x ds_load_b128, rotated double-buffer so LDS loads overlap
// the WMMA pipeline.
// ---------------------------------------------------------------------------
__global__ __launch_bounds__(256) void
proj_gemm_kernel(const unsigned short* __restrict__ hb,   // [N,H] bf16
                 const unsigned short* __restrict__ Wb4,  // 4 x [H,H] swizzled bf16
                 const float* __restrict__ b0, const float* __restrict__ b1,
                 const float* __restrict__ b2, const float* __restrict__ b3,
                 float* __restrict__ o0, float* __restrict__ o1,
                 float* __restrict__ o2, float* __restrict__ o3) {
    __shared__ unsigned short wtile[HH];   // 32 KB, fragment-major

    const int y = blockIdx.y;
    const unsigned short* Wmat = Wb4 + (size_t)y * HH;
    const float* bias = (y == 0) ? b0 : (y == 1) ? b1 : (y == 2) ? b2 : b3;
    float* out        = (y == 0) ? o0 : (y == 1) ? o1 : (y == 2) ? o2 : o3;

    // stage the 32 KB weight tile into LDS
#ifdef HAVE_ASYNC_LDS
    {
        const uint4* src = (const uint4*)Wmat;
        uint4* dst = (uint4*)wtile;
        for (int i = threadIdx.x; i < HH / 8; i += blockDim.x) {
            __builtin_amdgcn_global_load_async_to_lds_b128(
                (__attribute__((address_space(1))) v4i_vs*)(src + i),
                (__attribute__((address_space(3))) v4i_vs*)(dst + i),
                0, 0);
        }
#if __has_builtin(__builtin_amdgcn_s_wait_asynccnt)
        __builtin_amdgcn_s_wait_asynccnt(0);
#else
        asm volatile("s_wait_asynccnt 0x0" ::: "memory");
#endif
    }
#else
    {
        const uint4* src = (const uint4*)Wmat;
        uint4* dst = (uint4*)wtile;
        for (int i = threadIdx.x; i < HH / 8; i += blockDim.x) dst[i] = src[i];
    }
#endif
    __syncthreads();

    const int wave = threadIdx.x >> 5;
    const int lane = threadIdx.x & 31;
    const int mt   = blockIdx.x * 8 + wave;
    if (mt >= MTILES) return;

    const int hi = lane >> 4;        // 0: lanes 0-15, 1: lanes 16-31
    const int lo = lane & 15;

    // --- A fragments (16x32 bf16 per K-step), ISA lane layout:
    // lane<16 : row=lo, K={kk*32+0..7, kk*32+16..23}
    // lane>=16: row=lo, K={kk*32+8..15, kk*32+24..31}
    Frag16 afrag[4];
    {
        const int rowA = mt * 16 + lo;
        const uint4* arow = (const uint4*)(hb + (size_t)rowA * HDIM);
#pragma unroll
        for (int kk = 0; kk < 4; ++kk) {
            afrag[kk].q[0] = arow[kk * 4 + hi];
            afrag[kk].q[1] = arow[kk * 4 + hi + 2];
        }
    }

    const uint4* bw = (const uint4*)wtile;   // fragment-major: [frag][lane][2 x uint4]

    v8f acc[8];
#pragma unroll
    for (int nt = 0; nt < 8; ++nt)
        acc[nt] = (v8f){0.f, 0.f, 0.f, 0.f, 0.f, 0.f, 0.f, 0.f};

#pragma unroll
    for (int kk = 0; kk < 4; ++kk) {
        Frag16 bcur;
        {
            const int idx = (kk * 32 + lane) * 2;       // frag (nt=0, kk)
            bcur.q[0] = bw[idx];
            bcur.q[1] = bw[idx + 1];
        }
#pragma unroll
        for (int nt = 0; nt < 8; ++nt) {
            Frag16 bnext;
            if (nt < 7) {
                const int idx = (((nt + 1) * 4 + kk) * 32 + lane) * 2;
                bnext.q[0] = bw[idx];
                bnext.q[1] = bw[idx + 1];
            }
            acc[nt] = __builtin_amdgcn_wmma_f32_16x16x32_bf16(
                          false, afrag[kk].v, false, bcur.v, (short)0,
                          acc[nt], false, false);
            if (nt < 7) bcur = bnext;
        }
    }

    // D layout: VGPR r -> row r (lanes 0-15) / row 8+r (lanes 16-31)
#pragma unroll
    for (int nt = 0; nt < 8; ++nt) {
        const int col = nt * 16 + lo;
        const float bcol = bias[col];
        const int rbase = mt * 16 + (hi << 3);
#pragma unroll
        for (int r = 0; r < 8; ++r)
            out[(size_t)(rbase + r) * HDIM + col] = acc[nt][r] + bcol;
    }
}

// ---------------------------------------------------------------------------
// Edge phase: one wave32 per edge, lane handles float4 (32*4 = 128 = H).
// gate = sigmoid(k[dst] + q[src] + ea*We + be); s[dst] += gate * v[src]
// (s already holds h@Ws + bs, so the scatter-add lands on the skip term).
// ---------------------------------------------------------------------------
__global__ __launch_bounds__(256) void
edge_kernel(const float* __restrict__ k, const float* __restrict__ q,
            const float* __restrict__ v, float* __restrict__ s,
            const int* __restrict__ ei, const float* __restrict__ ea,
            const float* __restrict__ Wel, const float* __restrict__ bel) {
    const int e = (blockIdx.x * blockDim.x + threadIdx.x) >> 5;
    if (e >= NEDGES) return;
    const int lane = threadIdx.x & 31;
    const int c4 = lane * 4;

    const int src = ei[e];
    const int dst = ei[NEDGES + e];
    const float eav = ea[e];   // ED == 1

    const float4 kd  = *(const float4*)(k + (size_t)dst * HDIM + c4);
    const float4 qs  = *(const float4*)(q + (size_t)src * HDIM + c4);
    const float4 vs  = *(const float4*)(v + (size_t)src * HDIM + c4);
    const float4 we4 = *(const float4*)(Wel + c4);
    const float4 be4 = *(const float4*)(bel + c4);

    const float mx = sigm(kd.x + qs.x + eav * we4.x + be4.x) * vs.x;
    const float my = sigm(kd.y + qs.y + eav * we4.y + be4.y) * vs.y;
    const float mz = sigm(kd.z + qs.z + eav * we4.z + be4.z) * vs.z;
    const float mw = sigm(kd.w + qs.w + eav * we4.w + be4.w) * vs.w;

    float* sp = s + (size_t)dst * HDIM + c4;
    atomicAdd(sp + 0, mx);
    atomicAdd(sp + 1, my);
    atomicAdd(sp + 2, mz);
    atomicAdd(sp + 3, mw);
}

// ---------------------------------------------------------------------------
// Node update: LeakyReLU + eval BatchNorm, emit fp32 h and bf16 h.
// ---------------------------------------------------------------------------
__global__ __launch_bounds__(HDIM) void
node_kernel(const float* __restrict__ s,
            const float* __restrict__ gamma, const float* __restrict__ beta,
            const float* __restrict__ rmean, const float* __restrict__ rvar,
            float* __restrict__ hf, unsigned short* __restrict__ hb) {
    const int n = blockIdx.x;
    const int h = threadIdx.x;
    float t = s[(size_t)n * HDIM + h];
    t = (t >= 0.f) ? t : 0.01f * t;
    const float y = gamma[h] * (t - rmean[h]) * rsqrtf(rvar[h] + 1e-5f) + beta[h];
    hf[(size_t)n * HDIM + h] = y;
    hb[(size_t)n * HDIM + h] = f2bf_rne(y);
}

// ---------------------------------------------------------------------------
// Pooling: zero accumulators, atomic segment sums, then classifier.
// ---------------------------------------------------------------------------
__global__ void pool_zero_kernel(float* __restrict__ psum, float* __restrict__ pcnt) {
    int i = blockIdx.x * blockDim.x + threadIdx.x;
    if (i < NGRAPH * HDIM) psum[i] = 0.f;
    if (i < NGRAPH) pcnt[i] = 0.f;
}

__global__ __launch_bounds__(HDIM) void
pool_kernel(const float* __restrict__ hf, const int* __restrict__ batch,
            float* __restrict__ psum, float* __restrict__ pcnt) {
    const int n = blockIdx.x;
    const int h = threadIdx.x;
    const int g = batch[n];
    atomicAdd(&psum[(size_t)g * HDIM + h], hf[(size_t)n * HDIM + h]);
    if (h == 0) atomicAdd(&pcnt[g], 1.0f);
}

__global__ __launch_bounds__(NGRAPH * NCLASS) void
classifier_kernel(const float* __restrict__ psum, const float* __restrict__ pcnt,
                  const float* __restrict__ clinical,
                  const float* __restrict__ Wc, const float* __restrict__ bc,
                  float* __restrict__ out) {
    const int t = threadIdx.x;            // 128 threads: (g, c)
    const int g = t >> 1;
    const int c = t & 1;
    const float inv = 1.0f / fmaxf(pcnt[g], 1.0f);
    float acc = bc[c];
#pragma unroll 4
    for (int h = 0; h < HDIM; ++h)
        acc += (psum[(size_t)g * HDIM + h] * inv) * Wc[h * NCLASS + c];
#pragma unroll
    for (int j = 0; j < NCLIN; ++j)
        acc += clinical[g * NCLIN + j] * Wc[(HDIM + j) * NCLASS + c];
    out[g * NCLASS + c] = acc;
}

// ---------------------------------------------------------------------------
// Launch
// ---------------------------------------------------------------------------
extern "C" void kernel_launch(void* const* d_in, const int* in_sizes, int n_in,
                              void* d_out, int out_size, void* d_ws, size_t ws_size,
                              hipStream_t stream) {
    const float* x        = (const float*)d_in[0];
    const int*   ei       = (const int*)  d_in[1];
    const float* ea       = (const float*)d_in[2];
    const int*   batch    = (const int*)  d_in[3];
    const float* clinical = (const float*)d_in[4];
    const float* Wk = (const float*)d_in[5];
    const float* bk = (const float*)d_in[6];
    const float* Wq = (const float*)d_in[7];
    const float* bq = (const float*)d_in[8];
    const float* Wv = (const float*)d_in[9];
    const float* bv = (const float*)d_in[10];
    const float* Ws = (const float*)d_in[11];
    const float* bs = (const float*)d_in[12];
    const float* We = (const float*)d_in[13];
    const float* be = (const float*)d_in[14];
    const float* gamma = (const float*)d_in[15];
    const float* beta  = (const float*)d_in[16];
    const float* rmean = (const float*)d_in[17];
    const float* rvar  = (const float*)d_in[18];
    const float* Wc = (const float*)d_in[19];
    const float* bc = (const float*)d_in[20];
    float* out = (float*)d_out;

    // workspace carve-up
    char* p = (char*)d_ws;
    const size_t NH = (size_t)NNODES * HDIM;
    unsigned short* hb   = (unsigned short*)p; p += NH * sizeof(unsigned short); // 12.8 MB
    float* hf   = (float*)p; p += NH * sizeof(float);                            // 25.6 MB
    float* kbuf = (float*)p; p += NH * sizeof(float);
    float* qbuf = (float*)p; p += NH * sizeof(float);
    float* vbuf = (float*)p; p += NH * sizeof(float);
    float* sbuf = (float*)p; p += NH * sizeof(float);
    unsigned short* Wb = (unsigned short*)p; p += (size_t)3 * 4 * HH * sizeof(unsigned short);
    float* psum = (float*)p; p += (size_t)NGRAPH * HDIM * sizeof(float);
    float* pcnt = (float*)p; p += (size_t)NGRAPH * sizeof(float);
    (void)ws_size; (void)n_in; (void)in_sizes; (void)out_size;

    // weights -> bf16, pre-swizzled to WMMA fragment order
    {
        const int total = 3 * 4 * HH;
        cvt_weights_kernel<<<(total + 255) / 256, 256, 0, stream>>>(Wk, Wq, Wv, Ws, Wb);
    }
    // x -> bf16 activations
    cvt_f32_bf16_kernel<<<(int)((NH + 255) / 256), 256, 0, stream>>>(x, hb, (int)NH);

    const dim3 gemmGrid((MTILES + 7) / 8, 4);
    for (int l = 0; l < 3; ++l) {
        proj_gemm_kernel<<<gemmGrid, 256, 0, stream>>>(
            hb, Wb + (size_t)l * 4 * HH,
            bk + (size_t)l * HDIM, bq + (size_t)l * HDIM,
            bv + (size_t)l * HDIM, bs + (size_t)l * HDIM,
            kbuf, qbuf, vbuf, sbuf);

        edge_kernel<<<NEDGES / 8, 256, 0, stream>>>(
            kbuf, qbuf, vbuf, sbuf, ei, ea,
            We + (size_t)l * HDIM, be + (size_t)l * HDIM);

        node_kernel<<<NNODES, HDIM, 0, stream>>>(
            sbuf,
            gamma + (size_t)l * HDIM, beta + (size_t)l * HDIM,
            rmean + (size_t)l * HDIM, rvar + (size_t)l * HDIM,
            hf, hb);
    }

    pool_zero_kernel<<<(NGRAPH * HDIM + 255) / 256, 256, 0, stream>>>(psum, pcnt);
    pool_kernel<<<NNODES, HDIM, 0, stream>>>(hf, batch, psum, pcnt);
    classifier_kernel<<<1, NGRAPH * NCLASS, 0, stream>>>(psum, pcnt, clinical, Wc, bc, out);
}
